// VLDCNP_47081431499370
// MI455X (gfx1250) — matrix operand order repaired and checked by reading, not compiled, not run on hardware
//
#include <hip/hip_runtime.h>
#include <stdint.h>

// ---------------------------------------------------------------------------
// VLDCNP fused attention for MI455X (gfx1250)
// wave32, v_wmma_f32_16x16x32_bf16, GLOBAL_LOAD_ASYNC_TO_LDS, 300KB WGP LDS
// ---------------------------------------------------------------------------
#define TT_  65
#define H_   512
#define H3_  1536
#define LQ_  64
#define ROWS 32          // M-rows per block (two 16-row WMMA stripes)
#define LN_EPS 1e-12f

typedef __attribute__((ext_vector_type(16))) __bf16 v16bf;
typedef __attribute__((ext_vector_type(8)))  float  v8f;

// ---------------------------------------------------------------------------
// Kernel A: q = relu(txt_fea @ W_proj + b_proj) -> bf16 workspace, TRANSPOSED:
//   qt[b, d, l]  (d = 0..H3-1, l = 0..L-1)
// One wave32 per 16x16 output tile; K = 512 (16 wmma steps).
// ---------------------------------------------------------------------------
__global__ __launch_bounds__(32) void vldcnp_proj_relu(
    const float* __restrict__ txt,   // [B, LQ, H]
    const float* __restrict__ W,     // [H, H3]
    const float* __restrict__ bias,  // [H3]
    __bf16* __restrict__ qt)         // [B, H3, LQ] bf16 (transposed)
{
    const int b    = blockIdx.z;
    const int m0   = blockIdx.y * 16;   // l-tile
    const int n0   = blockIdx.x * 16;   // d-tile
    const int lane = threadIdx.x;
    const int lhi  = lane >> 4;
    const int llo  = lane & 15;

    const int arow  = m0 + llo;
    const int koffA = 16 * lhi;
    const int ncol  = n0 + llo;
    const int koffB = 16 * lhi;

    const float* trow = txt + ((size_t)b * LQ_ + arow) * H_;

    v8f acc = {};
    for (int kb = 0; kb < H_; kb += 32) {
        v16bf a, bm;
#pragma unroll
        for (int j = 0; j < 8; ++j) {
            a[j]     = (__bf16)trow[kb + koffA + j];        // K = kb+koffA+j
            a[j + 8] = (__bf16)trow[kb + 8 + koffA + j];    // K = kb+8+koffA+j
        }
#pragma unroll
        for (int j = 0; j < 16; ++j)
            bm[j] = (__bf16)W[(size_t)(kb + koffB + j) * H3_ + ncol];
        acc = __builtin_amdgcn_wmma_f32_16x16x32_bf16(
            false, a, false, bm, (short)0, acc, false, false);
    }

    const float bi = bias[ncol];
#pragma unroll
    for (int r = 0; r < 8; ++r) {
        int m = m0 + r + 8 * lhi;                       // l index
        float v = fmaxf(acc[r] + bi, 0.0f);
        qt[((size_t)b * H3_ + ncol) * LQ_ + m] = (__bf16)v;   // transposed store
    }
}

// ---------------------------------------------------------------------------
// Kernel B: fused gather -> scores -> softmax -> sv -> residual+LN -> scatter.
// Block = 256 threads (8 waves). Dynamic LDS ~300 KB (CDNA5 WGP: 320 KB).
// LDS layout (bytes):
//   [0      ) qt_lds : 1536*64 bf16   = 196608   (q transposed: [d][l])
//   [196608 ) v_lds  : 32*1536 bf16   =  98304
//   [294912 ) sc     : 32*64  f32     =   8192
//   [303104 ) p_lds  : 32*64  bf16    =   4096
//   [307200 ) stats  : 128 f32        =    512   (sum, sumsq, mu, rstd)
// ---------------------------------------------------------------------------
#define SMEM_BYTES 307712

__global__ __launch_bounds__(256) void vldcnp_main(
    const float* __restrict__ vis,     // [B, T*T, H3]
    const __bf16* __restrict__ qt,     // [B, H3, LQ] bf16
    const float* __restrict__ lnw,
    const float* __restrict__ lnb,
    const int* __restrict__ idx,       // [M]
    int M,
    float* __restrict__ out)           // [B, T*T, H3]
{
    extern __shared__ __align__(16) char smem[];
    __bf16* qt_lds = (__bf16*)(smem);
    __bf16* v_lds  = (__bf16*)(smem + 196608);
    float*  sc     = (float*)(smem + 294912);
    __bf16* p_lds  = (__bf16*)(smem + 303104);
    float*  stats  = (float*)(smem + 307200);

    const int b     = blockIdx.y;
    const int mbase = blockIdx.x * ROWS;
    const int tid   = threadIdx.x;
    const int wave  = tid >> 5;
    const int lane  = tid & 31;
    const int lhi   = lane >> 4;
    const int llo   = lane & 15;

    if (tid < 128) stats[tid] = 0.0f;

    // ---- stage q_t via gfx1250 async global->LDS copy (ASYNCcnt-tracked) ----
    {
        const __bf16* gsrc = qt + (size_t)b * H3_ * LQ_;
        const unsigned lds_base = (unsigned)(uintptr_t)(void*)qt_lds;
        for (int i = tid; i < (H3_ * LQ_) / 8; i += 256) {       // 16B chunks
            const void* g = (const void*)(gsrc + (size_t)i * 8);
            unsigned l = lds_base + (unsigned)i * 16u;
            asm volatile("global_load_async_to_lds_b128 %0, %1, off"
                         :: "v"(l), "v"(g) : "memory");
        }
        asm volatile("s_wait_asynccnt 0x0" ::: "memory");
    }

    // ---- gather v rows + convert to bf16: 8 threads per row ----
    {
        const int rr  = tid >> 3;
        const int sub = tid & 7;
        int m = mbase + rr; if (m >= M) m = M - 1;
        const float4* src = (const float4*)(vis + ((size_t)b * (TT_ * TT_) + idx[m]) * H3_);
        __bf16* dst = v_lds + rr * H3_;
        for (int c4 = sub; c4 < H3_ / 4; c4 += 8) {
            float4 f = src[c4];
            int c = c4 * 4;
            dst[c + 0] = (__bf16)f.x;
            dst[c + 1] = (__bf16)f.y;
            dst[c + 2] = (__bf16)f.z;
            dst[c + 3] = (__bf16)f.w;
        }
    }
    __syncthreads();

    // ---- phase 2: scores a[32,64] = v_tile . q^T  (8 waves -> 8 tiles) ----
    {
        const int stripe = wave & 1;       // 16-row stripe
        const int ct     = wave >> 1;      // 16-col tile of L
        const int arow   = stripe * 16 + llo;
        const int koffA  = 16 * lhi;
        const int ncol   = ct * 16 + llo;  // l index
        const int koffB  = 16 * lhi;
        v8f acc = {};
        for (int kb = 0; kb < H3_; kb += 32) {
            v16bf a, bm;
            const __bf16* ap = v_lds + arow * H3_ + kb + koffA;
#pragma unroll
            for (int j = 0; j < 8; ++j) { a[j] = ap[j]; a[j + 8] = ap[8 + j]; }
#pragma unroll
            for (int j = 0; j < 16; ++j)                    // B[k,l] = qt[k][l]
                bm[j] = qt_lds[(kb + koffB + j) * LQ_ + ncol];
            acc = __builtin_amdgcn_wmma_f32_16x16x32_bf16(
                false, a, false, bm, (short)0, acc, false, false);
        }
#pragma unroll
        for (int r = 0; r < 8; ++r) {
            int row = stripe * 16 + r + 8 * lhi;
            sc[row * LQ_ + ct * 16 + llo] = acc[r];
        }
    }
    __syncthreads();

    // ---- phase 3: softmax over L=64, one thread per row -> p_lds (bf16) ----
    if (tid < ROWS) {
        float* s = sc + tid * LQ_;
        float mx = -3.4e38f;
        for (int l = 0; l < LQ_; ++l) mx = fmaxf(mx, s[l]);
        float sum = 0.0f;
        for (int l = 0; l < LQ_; ++l) { float e = __expf(s[l] - mx); sum += e; s[l] = e; }
        float inv = 1.0f / sum;
        for (int l = 0; l < LQ_; ++l) p_lds[tid * LQ_ + l] = (__bf16)(s[l] * inv);
    }
    __syncthreads();

    // ---- phase 4 pass 1: sv tiles -> LN statistics only (no sv storage) ----
    const int stripe = wave & 1;
    const int koffA  = 16 * lhi;
    const int koffB  = 16 * lhi;
    const int prow   = stripe * 16 + llo;
    {
        float s1[8], s2[8];
#pragma unroll
        for (int r = 0; r < 8; ++r) { s1[r] = 0.0f; s2[r] = 0.0f; }
        for (int i = 0; i < 24; ++i) {
            const int ct  = (wave >> 1) + 4 * i;      // 0..95 column tile
            const int col = ct * 16 + llo;
            v8f acc = {};
#pragma unroll
            for (int kb = 0; kb < LQ_; kb += 32) {
                v16bf a, bm;
                const __bf16* ap = p_lds + prow * LQ_ + kb + koffA;
#pragma unroll
                for (int j = 0; j < 8; ++j) { a[j] = ap[j]; a[j + 8] = ap[8 + j]; }
                const __bf16* bp = qt_lds + col * LQ_ + kb + koffB;  // B[k,d]=qt[d][k]
#pragma unroll
                for (int j = 0; j < 16; ++j) bm[j] = bp[j];          // contiguous
                acc = __builtin_amdgcn_wmma_f32_16x16x32_bf16(
                    false, a, false, bm, (short)0, acc, false, false);
            }
#pragma unroll
            for (int r = 0; r < 8; ++r) {
                int rloc = stripe * 16 + r + 8 * lhi;
                int m = mbase + rloc; if (m >= M) m = M - 1;
                float x = acc[r] + vis[((size_t)b * (TT_ * TT_) + idx[m]) * H3_ + col];
                s1[r] += x; s2[r] += x * x;
            }
        }
#pragma unroll
        for (int r = 0; r < 8; ++r) {
            int rloc = stripe * 16 + r + 8 * lhi;
            atomicAdd(&stats[rloc],      s1[r]);   // ds_add_f32
            atomicAdd(&stats[32 + rloc], s2[r]);
        }
    }
    __syncthreads();

    if (tid < ROWS) {
        float mu  = stats[tid] * (1.0f / H3_);
        float var = stats[32 + tid] * (1.0f / H3_) - mu * mu;
        stats[64 + tid] = mu;
        stats[96 + tid] = rsqrtf(var + LN_EPS);
    }
    __syncthreads();

    // ---- phase 4 pass 2: recompute sv tiles, apply LayerNorm, scatter ----
    for (int i = 0; i < 24; ++i) {
        const int ct  = (wave >> 1) + 4 * i;
        const int col = ct * 16 + llo;
        v8f acc = {};
#pragma unroll
        for (int kb = 0; kb < LQ_; kb += 32) {
            v16bf a, bm;
            const __bf16* ap = p_lds + prow * LQ_ + kb + koffA;
#pragma unroll
            for (int j = 0; j < 8; ++j) { a[j] = ap[j]; a[j + 8] = ap[8 + j]; }
            const __bf16* bp = qt_lds + col * LQ_ + kb + koffB;
#pragma unroll
            for (int j = 0; j < 16; ++j) bm[j] = bp[j];
            acc = __builtin_amdgcn_wmma_f32_16x16x32_bf16(
                false, a, false, bm, (short)0, acc, false, false);
        }
        const float lw = lnw[col];
        const float lb = lnb[col];
#pragma unroll
        for (int r = 0; r < 8; ++r) {
            int rloc = stripe * 16 + r + 8 * lhi;
            int m = mbase + rloc;
            if (m < M) {
                size_t base = ((size_t)b * (TT_ * TT_) + idx[m]) * H3_ + col;
                float x = acc[r] + vis[base];
                out[base] = (x - stats[64 + rloc]) * stats[96 + rloc] * lw + lb;
            }
        }
    }
}

// ---------------------------------------------------------------------------
extern "C" void kernel_launch(void* const* d_in, const int* in_sizes, int n_in,
                              void* d_out, int out_size, void* d_ws, size_t ws_size,
                              hipStream_t stream) {
    const float* vis  = (const float*)d_in[0];
    const float* txt  = (const float*)d_in[1];
    const float* W    = (const float*)d_in[2];
    const float* bias = (const float*)d_in[3];
    const float* lnw  = (const float*)d_in[4];
    const float* lnb  = (const float*)d_in[5];
    const int*   idx  = (const int*)d_in[6];
    const int M = in_sizes[6];                           // 2145
    const int B = in_sizes[1] / (LQ_ * H_);              // 16
    float* out = (float*)d_out;
    __bf16* qt = (__bf16*)d_ws;                          // [B, H3, LQ] bf16

    // Scatter target is zeros everywhere except valid rows.
    hipMemsetAsync(d_out, 0, (size_t)out_size * sizeof(float), stream);

    // q = relu(txt @ W + b), stored transposed in bf16
    dim3 gA(H3_ / 16, LQ_ / 16, B);
    vldcnp_proj_relu<<<gA, 32, 0, stream>>>(txt, W, bias, qt);

    // fused attention + LN + scatter
    hipFuncSetAttribute((const void*)vldcnp_main,
                        hipFuncAttributeMaxDynamicSharedMemorySize, SMEM_BYTES);
    dim3 gB((M + ROWS - 1) / ROWS, B);
    vldcnp_main<<<gB, 256, SMEM_BYTES, stream>>>(vis, qt, lnw, lnb, idx, M, out);
}
